// RGCNEncoder_43645457662439
// MI455X (gfx1250) — matrix operand order repaired and checked by reading, not compiled, not run on hardware
//
#include <hip/hip_runtime.h>
#include <hip/hip_bf16.h>

// ---------------- problem constants ----------------
#define N_  100000
#define E_  640000
#define R_  16
#define B_  10
#define H_  128
#define L_  2
#define EPS_ 1e-5f

typedef __attribute__((ext_vector_type(16))) _Float16 v16h;
typedef __attribute__((ext_vector_type(8)))  _Float16 v8h;
typedef __attribute__((ext_vector_type(8)))  float    v8f;

// ---------------------------------------------------------------------------
// 1) per-(dst,rel) edge counts + per-relation counts
// ---------------------------------------------------------------------------
__global__ __launch_bounds__(256) void k_count(const int* __restrict__ dst,
                                               const int* __restrict__ et,
                                               int* __restrict__ cnt,
                                               int* __restrict__ relCnt) {
    int e = blockIdx.x * 256 + threadIdx.x;
    if (e >= E_) return;
    int r = et[e];
    atomicAdd(&cnt[dst[e] * R_ + r], 1);
    atomicAdd(&relCnt[r], 1);
}

// ---------------------------------------------------------------------------
// 2) tiny exclusive scan over 16 relation counts, padded to multiples of 32
//    (each wave consumes a relation-pure 32-edge group)
// ---------------------------------------------------------------------------
__global__ void k_scan(const int* __restrict__ relCnt,
                       int* __restrict__ workCnt,
                       int* __restrict__ padTotal) {
    if (threadIdx.x == 0 && blockIdx.x == 0) {
        int off = 0;
        for (int r = 0; r < R_; ++r) {
            workCnt[r] = off;
            off += ((relCnt[r] + 31) >> 5) << 5;   // pad each bucket to 32
        }
        padTotal[0] = off;                          // total padded edge slots
    }
}

// ---------------------------------------------------------------------------
// 3) bucket edge ids by relation (dummy slots stay -1 from memset 0xFF)
// ---------------------------------------------------------------------------
__global__ __launch_bounds__(256) void k_scatter(const int* __restrict__ et,
                                                 int* __restrict__ workCnt,
                                                 int* __restrict__ sorted) {
    int e = blockIdx.x * 256 + threadIdx.x;
    if (e >= E_) return;
    int p = atomicAdd(&workCnt[et[e]], 1);
    sorted[p] = e;
}

// ---------------------------------------------------------------------------
// 4) fold bases with comp into per-relation f16 weights, TRANSPOSED Wt[r][n][k]
//    slot r==16 holds transposed root matrix for this layer
// ---------------------------------------------------------------------------
__global__ __launch_bounds__(256) void k_weights(const float* __restrict__ bases,
                                                 const float* __restrict__ comp,
                                                 const float* __restrict__ root,
                                                 _Float16* __restrict__ Wt,
                                                 int l) {
    int idx = blockIdx.x * 256 + threadIdx.x;
    if (idx >= 17 * H_ * H_) return;
    int r  = idx / (H_ * H_);
    int nk = idx % (H_ * H_);
    int n  = nk / H_;
    int k  = nk % H_;
    float acc;
    if (r < R_) {
        acc = 0.f;
        #pragma unroll
        for (int b = 0; b < B_; ++b)
            acc += comp[(l * R_ + r) * B_ + b] *
                   bases[(((size_t)l * B_ + b) * H_ + k) * H_ + n];
    } else {
        acc = root[((size_t)l * H_ + k) * H_ + n];
    }
    Wt[((size_t)r * H_ + n) * H_ + k] = (_Float16)acc;
}

// ---------------------------------------------------------------------------
// 5) f32 -> f16 copy of the node features (layer 0 only)
// ---------------------------------------------------------------------------
__global__ __launch_bounds__(256) void k_x2h(const float* __restrict__ x,
                                             _Float16* __restrict__ xh) {
    size_t i = (size_t)blockIdx.x * 256 + threadIdx.x;
    if (i >= (size_t)N_ * H_) return;
    xh[i] = (_Float16)x[i];
}

// ---------------------------------------------------------------------------
// 6) message kernel: per relation-pure 32-edge group, each B fragment loaded
//    once and used by TWO WMMAs (rows 0-15 and 16-31), then mean-scaled f32
//    atomic scatter-add into agg
// ---------------------------------------------------------------------------
__global__ __launch_bounds__(128) void k_msg(const _Float16* __restrict__ xh,
                                             const _Float16* __restrict__ Wt,
                                             const int* __restrict__ sorted,
                                             const int* __restrict__ srcA,
                                             const int* __restrict__ dstA,
                                             const int* __restrict__ et,
                                             const int* __restrict__ cnt,
                                             const int* __restrict__ padTotal,
                                             float* __restrict__ agg) {
    __shared__ _Float16 As[4][32][H_];   // per-wave 32x128 f16 A tiles (8KB)
    __shared__ float    invS[4][32];
    __shared__ int      dstS[4][32];

    const int wid  = threadIdx.x >> 5;
    const int lane = threadIdx.x & 31;
    const int grp  = blockIdx.x * 4 + wid;
    const int total = padTotal[0];

    bool active = (grp * 32 < total);
    const int base = grp * 32;
    int r = 0;
    if (active) {
        int e0 = sorted[base];
        if (e0 < 0) active = false;       // cannot happen, defensive
        else        r = et[e0];           // group is relation-pure
    }

    if (active) {
        // metadata: destination + 1/cnt for all 32 rows (one per lane)
        {
            int e = sorted[base + lane];
            if (e >= 0) {
                int d = dstA[e];
                int c = cnt[d * R_ + r];
                invS[wid][lane] = 1.0f / (float)(c > 0 ? c : 1);
                dstS[wid][lane] = d;
            } else {
                invS[wid][lane] = 0.f;
                dstS[wid][lane] = -1;
            }
        }
        // gather 32 rows (256B each) of f16 features; 16B chunks across lanes
        for (int c = lane; c < 32 * 16; c += 32) {
            int m = c >> 4;
            int off = (c & 15) * 8;
            int e = sorted[base + m];
            uint4 val = make_uint4(0u, 0u, 0u, 0u);
            if (e >= 0)
                val = *(const uint4*)(xh + (size_t)srcA[e] * H_ + off);
            *(uint4*)(&As[wid][m][off]) = val;
        }
    }
    __syncthreads();

    if (active) {
        const int h2 = lane >> 4;     // lane half: selects K sub-ranges / M rows
        const int n  = lane & 15;     // output column within 16-wide N tile
        const int m  = lane & 15;     // A row within each 16-row tile
        const _Float16* wr = Wt + (size_t)r * H_ * H_;   // Wt[r][n][k]

        // preload all A fragments: 2 row-tiles x 4 K-steps (64 VGPRs)
        v16h a0[4], a1[4];
        #pragma unroll
        for (int kt = 0; kt < 4; ++kt) {
            v8h lo0 = *(const v8h*)(&As[wid][m][kt * 32 + h2 * 16]);
            v8h hi0 = *(const v8h*)(&As[wid][m][kt * 32 + h2 * 16 + 8]);
            v8h lo1 = *(const v8h*)(&As[wid][16 + m][kt * 32 + h2 * 16]);
            v8h hi1 = *(const v8h*)(&As[wid][16 + m][kt * 32 + h2 * 16 + 8]);
            #pragma unroll
            for (int i = 0; i < 8; ++i) {
                a0[kt][i] = lo0[i]; a0[kt][i + 8] = hi0[i];
                a1[kt][i] = lo1[i]; a1[kt][i + 8] = hi1[i];
            }
        }

        for (int nt = 0; nt < 8; ++nt) {
            v8f acc0 = {};
            v8f acc1 = {};
            const _Float16* bp = wr + (size_t)(nt * 16 + n) * H_;
            #pragma unroll
            for (int kt = 0; kt < 4; ++kt) {
                // B fragment loaded ONCE, consumed by two WMMAs
                v16h b = *(const v16h*)(bp + kt * 32 + h2 * 16);
                acc0 = __builtin_amdgcn_wmma_f32_16x16x32_f16(
                        false, a0[kt], false, b, (short)0, acc0, false, false);
                acc1 = __builtin_amdgcn_wmma_f32_16x16x32_f16(
                        false, a1[kt], false, b, (short)0, acc1, false, false);
            }
            // D layout: lane -> col n, VGPR v -> row v + 8*h2 (per 16-row tile)
            #pragma unroll
            for (int v = 0; v < 8; ++v) {
                int m0 = v + 8 * h2;
                int d0 = dstS[wid][m0];
                if (d0 >= 0)
                    atomicAdd(agg + (size_t)d0 * H_ + nt * 16 + n,
                              acc0[v] * invS[wid][m0]);
                int m1 = 16 + m0;
                int d1 = dstS[wid][m1];
                if (d1 >= 0)
                    atomicAdd(agg + (size_t)d1 * H_ + nt * 16 + n,
                              acc1[v] * invS[wid][m1]);
            }
        }
    }
}

// ---------------------------------------------------------------------------
// 7) root matmul (WMMA) + agg + bias, LayerNorm, ReLU, residual
//    writes f32 output and (optionally) f16 copy for the next layer
// ---------------------------------------------------------------------------
__global__ __launch_bounds__(128) void k_rootln(const float* __restrict__ xin,
                                                const _Float16* __restrict__ xh,
                                                const _Float16* __restrict__ Wr, // root, transposed
                                                const float* __restrict__ agg,
                                                const float* __restrict__ bias,
                                                const float* __restrict__ gamma,
                                                const float* __restrict__ beta,
                                                float* __restrict__ xout,
                                                _Float16* __restrict__ xhout,
                                                int writeHalf) {
    __shared__ _Float16 As[4][16][H_];
    __shared__ float    Os[4][16][H_];

    const int wid  = threadIdx.x >> 5;
    const int lane = threadIdx.x & 31;
    const int tile = blockIdx.x * 4 + wid;
    const bool active = (tile * 16 < N_);   // N_ % 16 == 0
    const int row0 = tile * 16;

    if (active) {
        for (int c = lane; c < 16 * 16; c += 32) {
            int m = c >> 4;
            int off = (c & 15) * 8;
            *(uint4*)(&As[wid][m][off]) =
                *(const uint4*)(xh + (size_t)(row0 + m) * H_ + off);
        }
    }
    __syncthreads();

    if (active) {
        const int h2 = lane >> 4;
        const int n  = lane & 15;
        const int m  = lane & 15;
        for (int nt = 0; nt < 8; ++nt) {
            v8f acc = {};
            const _Float16* bp = Wr + (size_t)(nt * 16 + n) * H_;
            #pragma unroll
            for (int kt = 0; kt < 4; ++kt) {
                v8h lo = *(const v8h*)(&As[wid][m][kt * 32 + h2 * 16]);
                v8h hi = *(const v8h*)(&As[wid][m][kt * 32 + h2 * 16 + 8]);
                v16h a;
                #pragma unroll
                for (int i = 0; i < 8; ++i) { a[i] = lo[i]; a[i + 8] = hi[i]; }
                v16h b = *(const v16h*)(bp + kt * 32 + h2 * 16);
                acc = __builtin_amdgcn_wmma_f32_16x16x32_f16(
                        false, a, false, b, (short)0, acc, false, false);
            }
            #pragma unroll
            for (int v = 0; v < 8; ++v)
                Os[wid][v + 8 * h2][nt * 16 + n] = acc[v];
        }
    }
    __syncthreads();

    if (active && lane < 16) {
        const int m = lane;
        const size_t node = (size_t)(row0 + m);
        float s = 0.f, s2 = 0.f;
        for (int c = 0; c < H_; ++c) {
            float h = Os[wid][m][c] + agg[node * H_ + c] + bias[c];
            Os[wid][m][c] = h;
            s  += h;
            s2 += h * h;
        }
        const float mu  = s * (1.0f / H_);
        const float var = s2 * (1.0f / H_) - mu * mu;
        const float rs  = rsqrtf(var + EPS_);
        for (int c = 0; c < H_; ++c) {
            float h = Os[wid][m][c];
            float y = gamma[c] * (h - mu) * rs + beta[c];
            y = fmaxf(y, 0.f) + xin[node * H_ + c];   // relu + residual
            xout[node * H_ + c] = y;
            if (writeHalf)
                xhout[node * H_ + c] = (_Float16)y;
        }
    }
}

// ---------------------------------------------------------------------------
// host launcher
// ---------------------------------------------------------------------------
extern "C" void kernel_launch(void* const* d_in, const int* in_sizes, int n_in,
                              void* d_out, int out_size, void* d_ws, size_t ws_size,
                              hipStream_t stream) {
    (void)in_sizes; (void)n_in; (void)out_size; (void)ws_size;

    const float* node_emb = (const float*)d_in[0];   // [N,H]
    const float* bases    = (const float*)d_in[1];   // [L,B,H,H]
    const float* comp     = (const float*)d_in[2];   // [L,R,B]
    const float* root     = (const float*)d_in[3];   // [L,H,H]
    const float* bias     = (const float*)d_in[4];   // [L,H]
    const float* gamma    = (const float*)d_in[5];   // [L,H]
    const float* beta     = (const float*)d_in[6];   // [L,H]
    const int*   eidx     = (const int*)d_in[7];     // [2,E]
    const int*   etype    = (const int*)d_in[8];     // [E]
    const int*   srcA = eidx;
    const int*   dstA = eidx + E_;
    float* out = (float*)d_out;                      // [N,H] f32

    // workspace carving (256B aligned)
    char* p = (char*)d_ws;
    auto carve = [&](size_t bytes) {
        char* q = p;
        p += (bytes + 255) & ~(size_t)255;
        return q;
    };
    int*      cnt      = (int*)carve((size_t)N_ * R_ * 4);        // 6.4 MB
    int*      relCnt   = (int*)carve(64);
    int*      workCnt  = (int*)carve(64);
    int*      padTotal = (int*)carve(64);
    int*      sorted   = (int*)carve((size_t)(E_ + 32 * R_) * 4); // 2.56 MB
    _Float16* Wt       = (_Float16*)carve((size_t)17 * H_ * H_ * 2);
    _Float16* xh       = (_Float16*)carve((size_t)N_ * H_ * 2);   // 25.6 MB
    float*    agg      = (float*)carve((size_t)N_ * H_ * 4);      // 51.2 MB

    // one-time bucketing by relation + counts
    hipMemsetAsync(cnt, 0, (size_t)N_ * R_ * 4, stream);
    hipMemsetAsync(relCnt, 0, 64, stream);
    hipMemsetAsync(sorted, 0xFF, (size_t)(E_ + 32 * R_) * 4, stream);
    k_count<<<(E_ + 255) / 256, 256, 0, stream>>>(dstA, etype, cnt, relCnt);
    k_scan<<<1, 1, 0, stream>>>(relCnt, workCnt, padTotal);
    k_scatter<<<(E_ + 255) / 256, 256, 0, stream>>>(etype, workCnt, sorted);
    k_x2h<<<(int)(((size_t)N_ * H_ + 255) / 256), 256, 0, stream>>>(node_emb, xh);

    const int maxGroups = E_ / 32 + R_;   // 20016 (upper bound on padded groups)
    const int nodeTiles = N_ / 16;        // 6250

    const float* xin = node_emb;
    for (int l = 0; l < L_; ++l) {
        k_weights<<<(17 * H_ * H_ + 255) / 256, 256, 0, stream>>>(
            bases, comp, root, Wt, l);
        hipMemsetAsync(agg, 0, (size_t)N_ * H_ * 4, stream);
        k_msg<<<(maxGroups + 3) / 4, 128, 0, stream>>>(
            xh, Wt, sorted, srcA, dstA, etype, cnt, padTotal, agg);
        k_rootln<<<(nodeTiles + 3) / 4, 128, 0, stream>>>(
            xin, xh, Wt + (size_t)16 * H_ * H_, agg,
            bias + l * H_, gamma + l * H_, beta + l * H_,
            out, xh, (l == 0) ? 1 : 0);
        xin = out;   // layer-0 output lives in d_out; layer 1 updates in place
    }
}
